// DGCNN_Reg_38680475467777
// MI455X (gfx1250) — matrix operand order repaired and checked by reading, not compiled, not run on hardware
//
#include <hip/hip_runtime.h>
#include <hip/hip_bf16.h>

typedef __attribute__((ext_vector_type(16))) __bf16 v16bf;
typedef __attribute__((ext_vector_type(8)))  float  v8f;
typedef unsigned short u16;
typedef unsigned int   u32;

#define WMMA_BF16(A, B, C) \
  __builtin_amdgcn_wmma_f32_16x16x32_bf16(false, (A), false, (B), (short)0, (C), false, false)

__device__ __forceinline__ float bf2f(u16 h) {
  return __uint_as_float(((u32)h) << 16);
}
__device__ __forceinline__ u16 f2bf(float f) {
  u32 u = __float_as_uint(f);
  u32 r = (u + 0x7FFFu + ((u >> 16) & 1u)) >> 16;   // round-to-nearest-even
  return (u16)r;
}
__device__ __forceinline__ v8f v8f_zero() {
  v8f z = {0.f,0.f,0.f,0.f,0.f,0.f,0.f,0.f};
  return z;
}
// A-fragment (16x32 bf16) K index for lane-half h, element e (contiguous: k = 16h + e)
__device__ __forceinline__ int a_kmap(int half, int e) {
  return e + 16*half;
}

// ---------------------------------------------------------------------------
// Pack x (B,1,N) f32 -> bf16 (B,N,32) channel 0 = x, rest zero.
__global__ __launch_bounds__(256) void pack_x0_kernel(const float* __restrict__ x,
                                                      u16* __restrict__ xp, int total) {
  int i = blockIdx.x * 256 + threadIdx.x;
  if (i < total) {
    u16* d = xp + (size_t)i * 32;
    d[0] = f2bf(x[i]);
    #pragma unroll
    for (int c = 1; c < 32; ++c) d[c] = 0;
  }
}

// ---------------------------------------------------------------------------
// Split-repack edge weights W (O x 2*Creal) f32 into TWO A-fragment bf16
// tensors over K = Cp only:  Wd = W[:, :C] * s   and   W2 = (W[:, C:] - W[:, :C]) * s
// (BN scale s folded in).  Then  W@[nbr-ctr;ctr] = Wd@nbr + W2@ctr.
__global__ __launch_bounds__(256) void repack_split_kernel(
    const float* __restrict__ W, const float* __restrict__ g, const float* __restrict__ rv,
    u16* __restrict__ WdF, u16* __restrict__ W2F, int O, int Creal, int Cp, int K2real) {
  int OT = O >> 4, KT = Cp >> 5;
  int ch = blockIdx.x * 256 + threadIdx.x;      // (rt*KT+kt)*32 + lane
  if (ch >= OT * KT * 32) return;
  int lane = ch & 31, rkt = ch >> 5;
  int kt = rkt % KT, rt = rkt / KT;
  int row = rt * 16 + (lane & 15), half = lane >> 4;
  float s = g[row] * rsqrtf(rv[row] + 1e-5f);
  u16 vd[16] __attribute__((aligned(32)));
  u16 v2[16] __attribute__((aligned(32)));
  #pragma unroll
  for (int e = 0; e < 16; ++e) {
    int K = kt * 32 + a_kmap(half, e);
    float wd = 0.f, w2 = 0.f;
    if (K < Creal) {
      wd = W[(size_t)row * K2real + K] * s;
      w2 = W[(size_t)row * K2real + Creal + K] * s - wd;
    }
    vd[e] = f2bf(wd);
    v2[e] = f2bf(w2);
  }
  ((int4*)&WdF[(size_t)ch * 16])[0] = ((int4*)vd)[0];
  ((int4*)&WdF[(size_t)ch * 16])[1] = ((int4*)vd)[1];
  ((int4*)&W2F[(size_t)ch * 16])[0] = ((int4*)v2)[0];
  ((int4*)&W2F[(size_t)ch * 16])[1] = ((int4*)v2)[1];
}

// ---------------------------------------------------------------------------
// Plain repack (used for w5, identity K mapping) with BN scale folded in.
__global__ __launch_bounds__(256) void repack_w_kernel(
    const float* __restrict__ W, const float* __restrict__ g, const float* __restrict__ rv,
    u16* __restrict__ Wf, int O, int Creal, int Cp, int K2real) {
  int OT = O >> 4, KT2 = Cp >> 4;
  int ch = blockIdx.x * 256 + threadIdx.x;
  if (ch >= OT * KT2 * 32) return;
  int lane = ch & 31, rkt = ch >> 5;
  int kt = rkt % KT2, rt = rkt / KT2;
  int row = rt * 16 + (lane & 15), half = lane >> 4;
  float s = g[row] * rsqrtf(rv[row] + 1e-5f);
  u16 vals[16] __attribute__((aligned(32)));
  #pragma unroll
  for (int e = 0; e < 16; ++e) {
    int K = kt * 32 + a_kmap(half, e);
    int c; bool ok;
    if (K < Cp) { c = K;                ok = (K < Creal); }
    else        { c = Creal + (K - Cp); ok = ((K - Cp) < Creal); }
    vals[e] = f2bf(ok ? W[(size_t)row * K2real + c] * s : 0.f);
  }
  ((int4*)&Wf[(size_t)ch * 16])[0] = ((int4*)vals)[0];
  ((int4*)&Wf[(size_t)ch * 16])[1] = ((int4*)vals)[1];
}

// ---------------------------------------------------------------------------
// Squared norms of each packed point row.
__global__ __launch_bounds__(256) void norms_kernel(const u16* __restrict__ Xin,
                                                    int stride, int off, int Cp,
                                                    float* __restrict__ out, int total) {
  int i = blockIdx.x * 256 + threadIdx.x;
  if (i < total) {
    const u16* p = Xin + (size_t)i * stride + off;
    float s = 0.f;
    for (int c = 0; c < Cp; ++c) { float v = bf2f(p[c]); s += v * v; }
    out[i] = s;
  }
}

// ---------------------------------------------------------------------------
// KNN: 16 queries per workgroup; 8 waves compute 16x16 Gram tiles via WMMA
// (B-fragments are contiguous 32B global loads of the channel-packed rows);
// 16 selector threads keep register top-20 lists (set semantics: max-over-k
// downstream is permutation invariant; self point pd=0 is naturally included).
__global__ __launch_bounds__(256) void knn_kernel(
    const u16* __restrict__ Xin, int strideIn, int offIn, int Cp,
    const float* __restrict__ norms, int* __restrict__ idxOut, int N) {
  __shared__ __attribute__((aligned(32))) u16 Af[4 * 32 * 16];   // KTC<=4 A fragments
  __shared__ float Pd[16 * 128];
  const int b = blockIdx.y, q0 = blockIdx.x * 16;
  const int tid = threadIdx.x, lane = tid & 31, wave = tid >> 5;
  const int KTC = Cp >> 5;

  for (int ch = tid; ch < KTC * 32; ch += 256) {
    int kt = ch >> 5, l = ch & 31;
    int row = l & 15, half = l >> 4;
    const u16* src = Xin + (size_t)(b * N + q0 + row) * strideIn + offIn + kt * 32;
    u16 vals[16] __attribute__((aligned(32)));
    #pragma unroll
    for (int e = 0; e < 16; ++e) vals[e] = src[a_kmap(half, e)];
    ((int4*)&Af[ch * 16])[0] = ((int4*)vals)[0];
    ((int4*)&Af[ch * 16])[1] = ((int4*)vals)[1];
  }
  __syncthreads();

  float dv[20]; int di[20];
  float minv = -3.4e38f; int minp = 0;
  #pragma unroll
  for (int i = 0; i < 20; ++i) { dv[i] = -3.4e38f; di[i] = 0; }

  for (int ci = 0; ci < N / 128; ++ci) {
    int cb0 = ci * 128 + wave * 16;
    v8f acc = v8f_zero();
    for (int kt = 0; kt < KTC; ++kt) {
      v16bf A = *(const v16bf*)&Af[(kt * 32 + lane) * 16];
      const u16* bp = Xin + (size_t)(b * N + cb0 + (lane & 15)) * strideIn
                      + offIn + kt * 32 + (lane >> 4) * 16;
      v16bf Bm = *(const v16bf*)bp;
      acc = WMMA_BF16(A, Bm, acc);
    }
    int col = cb0 + (lane & 15);
    float xc = norms[(size_t)b * N + col];
    #pragma unroll
    for (int r = 0; r < 8; ++r) {
      int rg = (lane >> 4) * 8 + r;
      float pd = 2.f * acc[r] - norms[(size_t)b * N + q0 + rg] - xc;
      Pd[rg * 128 + wave * 16 + (lane & 15)] = pd;
    }
    __syncthreads();
    if (tid < 16) {
      for (int j = 0; j < 128; ++j) {
        float v = Pd[tid * 128 + j];
        if (v > minv) {
          dv[minp] = v; di[minp] = ci * 128 + j;
          minv = dv[0]; minp = 0;
          #pragma unroll
          for (int i = 1; i < 20; ++i) if (dv[i] < minv) { minv = dv[i]; minp = i; }
        }
      }
    }
    __syncthreads();
  }
  if (tid < 16) {
    int* o = idxOut + ((size_t)b * N + q0 + tid) * 20;
    #pragma unroll
    for (int i = 0; i < 20; ++i) o[i] = di[i];
  }
}

// ---------------------------------------------------------------------------
// Fused EdgeConv with split weights:
//   y[o,p,kk] = (Wd @ x_nbr)[o] + (W2 @ x_ctr)[o,p] (+bias) -> LReLU -> max_kk
// Workgroup: one batch, 8 points (160 neighbor columns = 10 col-tiles + one
// 16-col center tile).  B-fragments load straight from the channel-packed
// activations (contiguous 32B per lane) -- no staging pass, K = Cp only.
template <int NRT>
__global__ __launch_bounds__(256) void edge_conv_kernel(
    const u16* __restrict__ Xin, int strideIn, int offIn, int Cp,
    const int* __restrict__ idx,
    const u16* __restrict__ WdF, const u16* __restrict__ W2F,
    const float* __restrict__ g, const float* __restrict__ bb,
    const float* __restrict__ rm, const float* __restrict__ rv,
    u16* __restrict__ Xout, int strideOut, int offOut, int O, int N) {
  constexpr int PTS = 8, KNB = 20, COLS = PTS * KNB, CT = COLS / 16; // 160, 10
  __shared__ float bias_s[256];
  __shared__ int   idx_s[COLS];
  __shared__ float T2s[256 * PTS];   // center term per (row, point), 8 KB
  __shared__ u16   Ys[128 * COLS];   // y tiles for max reduction, 40 KB

  const int wgPerB = N / PTS;
  const int b = blockIdx.x / wgPerB;
  const int p0 = (blockIdx.x % wgPerB) * PTS;
  const int tid = threadIdx.x, lane = tid & 31, wave = tid >> 5;
  const int KT = Cp >> 5;
  const int OT = O >> 4;
  const int colc = lane & 15, khalf = (lane >> 4) * 16;

  for (int o = tid; o < O; o += 256) {
    float s = g[o] * rsqrtf(rv[o] + 1e-5f);
    bias_s[o] = bb[o] - rm[o] * s;
  }
  for (int t = tid; t < COLS; t += 256)
    idx_s[t] = idx[((size_t)b * N + p0 + t / KNB) * KNB + t % KNB];
  __syncthreads();

  // Per-lane gather offsets (elements) + prefetch of the scattered rows.
  u32 boff[CT];
  #pragma unroll
  for (int ctc = 0; ctc < CT; ++ctc) {
    int nbr = idx_s[ctc * 16 + colc];
    boff[ctc] = (u32)(b * N + nbr) * strideIn + offIn;
    __builtin_prefetch(Xin + boff[ctc], 0, 0);
  }
  u32 coff = (u32)(b * N + p0 + (colc & 7)) * strideIn + offIn;

  v8f acc1[NRT][CT];
  v8f acc2[NRT];
  #pragma unroll
  for (int i = 0; i < NRT; ++i) {
    acc2[i] = v8f_zero();
    #pragma unroll
    for (int j = 0; j < CT; ++j) acc1[i][j] = v8f_zero();
  }

  if (wave < OT) {                    // wave-uniform; NRT=2 only used when OT=16
    for (int kt = 0; kt < KT; ++kt) {
      v16bf Ad[NRT], A2[NRT];
      #pragma unroll
      for (int nr = 0; nr < NRT; ++nr) {
        int rt = wave + nr * 8;
        if (rt < OT) {
          Ad[nr] = *(const v16bf*)&WdF[((size_t)(rt * KT + kt) * 32 + lane) * 16];
          A2[nr] = *(const v16bf*)&W2F[((size_t)(rt * KT + kt) * 32 + lane) * 16];
        }
      }
      #pragma unroll
      for (int ctc = 0; ctc < CT; ++ctc) {
        v16bf Bm = *(const v16bf*)(Xin + boff[ctc] + kt * 32 + khalf);
        #pragma unroll
        for (int nr = 0; nr < NRT; ++nr)
          if (wave + nr * 8 < OT)
            acc1[nr][ctc] = WMMA_BF16(Ad[nr], Bm, acc1[nr][ctc]);
      }
      v16bf Bc = *(const v16bf*)(Xin + coff + kt * 32 + khalf);
      #pragma unroll
      for (int nr = 0; nr < NRT; ++nr)
        if (wave + nr * 8 < OT)
          acc2[nr] = WMMA_BF16(A2[nr], Bc, acc2[nr]);
    }
  }

  // Spill broadcast center term to LDS.
  #pragma unroll
  for (int nr = 0; nr < NRT; ++nr) {
    int rt = wave + nr * 8;
    if (rt < OT && colc < 8) {
      int rsub = (lane >> 4) * 8;
      #pragma unroll
      for (int r = 0; r < 8; ++r)
        T2s[(rt * 16 + rsub + r) * PTS + colc] = acc2[nr][r];
    }
  }
  __syncthreads();

  // Epilogue: + center term + bias, LReLU -> LDS, max over k -> bf16 output.
  #pragma unroll
  for (int nr = 0; nr < NRT; ++nr) {
    int rt = wave + nr * 8;
    if (rt < OT) {
      int rsub = (lane >> 4) * 8;
      #pragma unroll
      for (int ctc = 0; ctc < CT; ++ctc) {
        int p = (ctc * 16 + colc) / KNB;
        #pragma unroll
        for (int r = 0; r < 8; ++r) {
          int row = rt * 16 + rsub + r;
          float y = acc1[nr][ctc][r] + T2s[row * PTS + p] + bias_s[row];
          y = y > 0.f ? y : 0.2f * y;
          Ys[(wave * 16 + rsub + r) * COLS + ctc * 16 + colc] = f2bf(y);
        }
      }
    }
    __syncthreads();
    int nwaves = OT - nr * 8; if (nwaves > 8) nwaves = 8; if (nwaves < 0) nwaves = 0;
    int nrows = nwaves * 16;
    for (int t = tid; t < nrows * PTS; t += 256) {
      int rloc = t / PTS, p = t % PTS;
      int row = rloc + nr * 128;
      float m = -3.4e38f;
      #pragma unroll
      for (int kk = 0; kk < KNB; ++kk)
        m = fmaxf(m, bf2f(Ys[rloc * COLS + p * KNB + kk]));
      Xout[(size_t)(b * N + p0 + p) * strideOut + offOut + row] = f2bf(m);
    }
    __syncthreads();
  }
}

// ---------------------------------------------------------------------------
// Final: y5 = LReLU(BN(w5 @ h)) then out = wreg . y5, fused.
__global__ __launch_bounds__(256) void final_reg_kernel(
    const u16* __restrict__ H, const u16* __restrict__ W5f,
    const float* __restrict__ wreg,
    const float* __restrict__ g, const float* __restrict__ bb,
    const float* __restrict__ rm, const float* __restrict__ rv,
    float* __restrict__ out, int N) {
  __shared__ float bias_s[1024];
  __shared__ float outs[16];
  const int b = blockIdx.y, n0 = blockIdx.x * 16;
  const int tid = threadIdx.x, lane = tid & 31, wave = tid >> 5;
  constexpr int KT2 = 16, NRT = 8;

  for (int o = tid; o < 1024; o += 256) {
    float s = g[o] * rsqrtf(rv[o] + 1e-5f);
    bias_s[o] = bb[o] - rm[o] * s;
  }
  if (tid < 16) outs[tid] = 0.f;
  __syncthreads();

  v8f acc[NRT];
  #pragma unroll
  for (int i = 0; i < NRT; ++i) acc[i] = v8f_zero();

  for (int kt = 0; kt < KT2; ++kt) {
    const u16* bp = H + (size_t)((size_t)b * N + n0 + (lane & 15)) * 512
                    + kt * 32 + (lane >> 4) * 16;
    v16bf Bm = *(const v16bf*)bp;
    #pragma unroll
    for (int nr = 0; nr < NRT; ++nr) {
      int rt = wave * NRT + nr;
      v16bf A = *(const v16bf*)&W5f[((size_t)(rt * KT2 + kt) * 32 + lane) * 16];
      acc[nr] = WMMA_BF16(A, Bm, acc[nr]);
    }
  }

  float partial = 0.f;
  #pragma unroll
  for (int nr = 0; nr < NRT; ++nr) {
    int rt = wave * NRT + nr;
    #pragma unroll
    for (int r = 0; r < 8; ++r) {
      int row = rt * 16 + (lane >> 4) * 8 + r;
      float y = acc[nr][r] + bias_s[row];
      y = y > 0.f ? y : 0.2f * y;
      partial += y * wreg[row];
    }
  }
  atomicAdd(&outs[lane & 15], partial);
  __syncthreads();
  if (tid < 16) out[(size_t)b * N + n0 + tid] = outs[tid];
}

// ---------------------------------------------------------------------------
extern "C" void kernel_launch(void* const* d_in, const int* in_sizes, int n_in,
                              void* d_out, int out_size, void* d_ws, size_t ws_size,
                              hipStream_t stream) {
  const int B = 8, N = 4096;
  const float* x    = (const float*)d_in[0];
  const float* w1   = (const float*)d_in[1];
  const float* w2   = (const float*)d_in[2];
  const float* w3   = (const float*)d_in[3];
  const float* w4   = (const float*)d_in[4];
  const float* w5   = (const float*)d_in[5];
  const float* wreg = (const float*)d_in[6];
  const float* g1 = (const float*)d_in[7],  *b1 = (const float*)d_in[8],
             *rm1 = (const float*)d_in[9],  *rv1 = (const float*)d_in[10];
  const float* g2 = (const float*)d_in[11], *b2 = (const float*)d_in[12],
             *rm2 = (const float*)d_in[13], *rv2 = (const float*)d_in[14];
  const float* g3 = (const float*)d_in[15], *b3 = (const float*)d_in[16],
             *rm3 = (const float*)d_in[17], *rv3 = (const float*)d_in[18];
  const float* g4 = (const float*)d_in[19], *b4 = (const float*)d_in[20],
             *rm4 = (const float*)d_in[21], *rv4 = (const float*)d_in[22];
  const float* g5 = (const float*)d_in[23], *b5 = (const float*)d_in[24],
             *rm5 = (const float*)d_in[25], *rv5 = (const float*)d_in[26];

  char* ws = (char*)d_ws;
  size_t off = 0;
  auto alloc = [&](size_t bytes) { size_t o = off; off += (bytes + 255) & ~(size_t)255; return o; };
  u16*   x0p  = (u16*)  (ws + alloc((size_t)B * N * 32 * 2));
  u16*   Hh   = (u16*)  (ws + alloc((size_t)B * N * 512 * 2));
  int*   idxb = (int*)  (ws + alloc((size_t)B * N * 20 * 4));
  float* nrms = (float*)(ws + alloc((size_t)B * N * 4));
  u16*   wd1  = (u16*)  (ws + alloc((size_t)4  * 1 * 32 * 16 * 2));
  u16*   w21  = (u16*)  (ws + alloc((size_t)4  * 1 * 32 * 16 * 2));
  u16*   wd2  = (u16*)  (ws + alloc((size_t)4  * 2 * 32 * 16 * 2));
  u16*   w22  = (u16*)  (ws + alloc((size_t)4  * 2 * 32 * 16 * 2));
  u16*   wd3  = (u16*)  (ws + alloc((size_t)8  * 2 * 32 * 16 * 2));
  u16*   w23  = (u16*)  (ws + alloc((size_t)8  * 2 * 32 * 16 * 2));
  u16*   wd4  = (u16*)  (ws + alloc((size_t)16 * 4 * 32 * 16 * 2));
  u16*   w24  = (u16*)  (ws + alloc((size_t)16 * 4 * 32 * 16 * 2));
  u16*   w5f  = (u16*)  (ws + alloc((size_t)64 * 16 * 32 * 16 * 2));

  const int total = B * N;
  pack_x0_kernel<<<(total + 255) / 256, 256, 0, stream>>>(x, x0p, total);

  // split repack for edge weights: (W, g, rv, Wd, W2, O, Creal, Cp, K2real)
  repack_split_kernel<<<1, 256, 0, stream>>>(w1, g1, rv1, wd1, w21,  64,   1,  32,   2);
  repack_split_kernel<<<1, 256, 0, stream>>>(w2, g2, rv2, wd2, w22,  64,  64,  64, 128);
  repack_split_kernel<<<2, 256, 0, stream>>>(w3, g3, rv3, wd3, w23, 128,  64,  64, 128);
  repack_split_kernel<<<8, 256, 0, stream>>>(w4, g4, rv4, wd4, w24, 256, 128, 128, 256);
  repack_w_kernel<<<(64 * 16 * 32 + 255) / 256, 256, 0, stream>>>(w5, g5, rv5, w5f,
                                                                  1024, 256, 256, 512);

  dim3 knnGrid(N / 16, B);
  int  ecGrid = B * (N / 8);

  // Block 1: x0p (Cp=32) -> H[:, 0:64)
  norms_kernel<<<(total + 255) / 256, 256, 0, stream>>>(x0p, 32, 0, 32, nrms, total);
  knn_kernel<<<knnGrid, 256, 0, stream>>>(x0p, 32, 0, 32, nrms, idxb, N);
  edge_conv_kernel<1><<<ecGrid, 256, 0, stream>>>(x0p, 32, 0, 32, idxb, wd1, w21,
                                                  g1, b1, rm1, rv1, Hh, 512, 0, 64, N);
  // Block 2: H[0:64) -> H[64:128)
  norms_kernel<<<(total + 255) / 256, 256, 0, stream>>>(Hh, 512, 0, 64, nrms, total);
  knn_kernel<<<knnGrid, 256, 0, stream>>>(Hh, 512, 0, 64, nrms, idxb, N);
  edge_conv_kernel<1><<<ecGrid, 256, 0, stream>>>(Hh, 512, 0, 64, idxb, wd2, w22,
                                                  g2, b2, rm2, rv2, Hh, 512, 64, 64, N);
  // Block 3: H[64:128) -> H[128:256)
  norms_kernel<<<(total + 255) / 256, 256, 0, stream>>>(Hh, 512, 64, 64, nrms, total);
  knn_kernel<<<knnGrid, 256, 0, stream>>>(Hh, 512, 64, 64, nrms, idxb, N);
  edge_conv_kernel<1><<<ecGrid, 256, 0, stream>>>(Hh, 512, 64, 64, idxb, wd3, w23,
                                                  g3, b3, rm3, rv3, Hh, 512, 128, 128, N);
  // Block 4: H[128:256) -> H[256:512)
  norms_kernel<<<(total + 255) / 256, 256, 0, stream>>>(Hh, 512, 128, 128, nrms, total);
  knn_kernel<<<knnGrid, 256, 0, stream>>>(Hh, 512, 128, 128, nrms, idxb, N);
  edge_conv_kernel<2><<<ecGrid, 256, 0, stream>>>(Hh, 512, 128, 128, idxb, wd4, w24,
                                                  g4, b4, rm4, rv4, Hh, 512, 256, 256, N);
  // Final fused GEMM + regression head
  final_reg_kernel<<<knnGrid, 256, 0, stream>>>(Hh, w5f, wreg, g5, b5, rm5, rv5,
                                                (float*)d_out, N);
  (void)in_sizes; (void)n_in; (void)out_size; (void)ws_size;
}
